// HDSymoblicAttention_63900523430276
// MI455X (gfx1250) — 8-bit matrix operand order repaired and checked by reading, compiled, not run on hardware
//
#include <hip/hip_runtime.h>
#include <hip/hip_bf16.h>
#include <math.h>

#define B_   16
#define S_   2048
#define DIN_ 64
#define DM_  64

typedef __attribute__((ext_vector_type(8))) int v8i;

#if __has_builtin(__builtin_amdgcn_exp2f)
#define EXP2F(x) __builtin_amdgcn_exp2f(x)
#else
#define EXP2F(x) exp2f(x)
#endif

__global__ void hdsym_zero(float* __restrict__ colsum) {
    int i = blockIdx.x * blockDim.x + threadIdx.x;
    if (i < B_ * DM_) colsum[i] = 0.f;
}

__global__ __launch_bounds__(256) void hdsym_phase1(
    const float* __restrict__ values, const float* __restrict__ W,
    const float* __restrict__ bias, const float* __restrict__ symbols,
    signed char* __restrict__ sg, float* __restrict__ sp,
    float* __restrict__ colsum) {
    __shared__ float swLds[DIN_ * DM_];
    __shared__ float vLds[B_ * DIN_];
    __shared__ float symLds[DIN_];

    const int s = blockIdx.x;
    const int t = threadIdx.x;
    const float* Wg = W + (size_t)s * DIN_ * DM_;

    #pragma unroll
    for (int i = 0; i < 16; ++i) {
        int idx = t + i * 256;
        float w = Wg[idx];
        swLds[idx] = (w > 0.f) ? 1.f : ((w < 0.f) ? -1.f : 0.f);
    }
    #pragma unroll
    for (int i = 0; i < 4; ++i) {
        int idx = t + i * 256;
        int b = idx >> 6, d = idx & 63;
        vLds[idx] = values[(size_t)b * S_ * DIN_ + (size_t)s * DIN_ + d];
    }
    if (t < DIN_) symLds[t] = symbols[(size_t)s * DIN_ + t];
    __syncthreads();

    #pragma unroll
    for (int i = 0; i < 4; ++i) {
        int o = t + i * 256;
        int b = o >> 6, e = o & 63;
        float y = bias[(size_t)s * DM_ + e];
        const float* vr = vLds + b * DIN_;
        #pragma unroll 8
        for (int d = 0; d < DIN_; ++d) y += vr[d] * swLds[d * DM_ + e];
        float vp = tanhf(y);
        sg[((size_t)b * S_ + s) * DM_ + e] =
            (vp > 0.f) ? (signed char)1 : ((vp < 0.f) ? (signed char)-1 : (signed char)0);
        atomicAdd(&colsum[b * DM_ + e], vp);
    }
    if (t < DM_) {
        float y = bias[(size_t)s * DM_ + t];
        #pragma unroll 8
        for (int d = 0; d < DIN_; ++d) y += symLds[d] * swLds[d * DM_ + t];
        sp[(size_t)s * DM_ + t] = tanhf(y);
    }
}

__global__ __launch_bounds__(32) void hdsym_phase2(
    const signed char* __restrict__ sg, float* __restrict__ diag) {
    const int b     = blockIdx.y;
    const int itile = blockIdx.x;
    const int lane  = threadIdx.x;
    const int n     = lane & 15;
    const int half  = lane >> 4;
    const float kScale = 1.4426950408889634f / 64.f;
    const int* sgw = (const int*)(sg + (size_t)b * S_ * DM_);

    const int* arow = sgw + (size_t)(itile * 16 + n) * 16;
    v8i a;
    #pragma unroll
    for (int p = 0; p < 2; ++p) {
        int4 aa = *(const int4*)(arow + p * 8 + half * 4);
        a[4 * p]     = aa.x;
        a[4 * p + 1] = aa.y;
        a[4 * p + 2] = aa.z;
        a[4 * p + 3] = aa.w;
    }

    auto loadB = [&](int jt) -> v8i {
        const int* brow = sgw + (size_t)(jt * 16 + n) * 16;
        int4 lo = *(const int4*)(brow + half * 4);
        int4 hi = *(const int4*)(brow + 8 + half * 4);
        v8i bm;
        bm[0] = lo.x; bm[1] = lo.y; bm[2] = lo.z; bm[3] = lo.w;
        bm[4] = hi.x; bm[5] = hi.y; bm[6] = hi.z; bm[7] = hi.w;
        return bm;
    };

    float acc0[8], acc1[8];
    #pragma unroll
    for (int r = 0; r < 8; ++r) { acc0[r] = 0.f; acc1[r] = 0.f; }

    #define WMMA_IU8(bm) __builtin_amdgcn_wmma_i32_16x16x64_iu8(true, a, true, (bm), c, false, false)
    #define CONSUME(dreg, acc)                                   \
        _Pragma("unroll")                                        \
        for (int r = 0; r < 8; ++r)                              \
            (acc)[r] += EXP2F((float)(dreg)[r] * kScale);

    const v8i c = {};
    v8i b0 = loadB(0), b1 = loadB(1), b2 = loadB(2), b3 = loadB(3);
    v8i d0 = WMMA_IU8(b0);
    v8i d1 = WMMA_IU8(b1);
    v8i d2 = WMMA_IU8(b2);
    v8i d3 = WMMA_IU8(b3);

    for (int jt = 4; jt < S_ / 16; jt += 4) {
        __builtin_prefetch(sgw + (size_t)((jt + 4) * 16 + n) * 16, 0, 1);
        v8i nb0 = loadB(jt);
        v8i nb1 = loadB(jt + 1);
        v8i nb2 = loadB(jt + 2);
        v8i nb3 = loadB(jt + 3);
        v8i e0 = WMMA_IU8(nb0);
        v8i e1 = WMMA_IU8(nb1);
        CONSUME(d0, acc0)
        CONSUME(d1, acc1)
        v8i e2 = WMMA_IU8(nb2);
        v8i e3 = WMMA_IU8(nb3);
        CONSUME(d2, acc0)
        CONSUME(d3, acc1)
        d0 = e0; d1 = e1; d2 = e2; d3 = e3;
    }
    CONSUME(d0, acc0)
    CONSUME(d1, acc1)
    CONSUME(d2, acc0)
    CONSUME(d3, acc1)
    #undef WMMA_IU8
    #undef CONSUME

    #pragma unroll
    for (int r = 0; r < 8; ++r) {
        float v = acc0[r] + acc1[r];
        #pragma unroll
        for (int m = 1; m < 16; m <<= 1) v += __shfl_xor(v, m, 32);
        if (n == 0)
            diag[(size_t)b * S_ + itile * 16 + r + 8 * half] =
                2.71828182845904523f / v;
    }
}

__global__ __launch_bounds__(256) void hdsym_phase3(
    const float* __restrict__ diag, const float* __restrict__ colsum,
    const float* __restrict__ sp, float* __restrict__ out) {
    int idx = blockIdx.x * blockDim.x + threadIdx.x;
    if (idx >= B_ * S_ * DM_) return;
    int e = idx & 63;
    int s = (idx >> 6) & (S_ - 1);
    int b = idx >> 17;
    float x = diag[b * S_ + s] * colsum[b * DM_ + e] * sp[s * DM_ + e];
    out[idx] = x / (1.f + __expf(-x));
}

extern "C" void kernel_launch(void* const* d_in, const int* in_sizes, int n_in,
                              void* d_out, int out_size, void* d_ws, size_t ws_size,
                              hipStream_t stream) {
    (void)in_sizes; (void)n_in; (void)out_size; (void)ws_size;
    const float* values  = (const float*)d_in[0];
    const float* W       = (const float*)d_in[1];
    const float* bias    = (const float*)d_in[2];
    const float* symbols = (const float*)d_in[3];
    float* out = (float*)d_out;

    char* ws = (char*)d_ws;
    signed char* sg = (signed char*)(ws);
    float* sp       = (float*)(ws + 2097152);
    float* diag     = (float*)(ws + 2097152 + 524288);
    float* colsum   = (float*)(ws + 2097152 + 524288 + 131072);

    hdsym_zero<<<4, 256, 0, stream>>>(colsum);
    hdsym_phase1<<<S_, 256, 0, stream>>>(values, W, bias, symbols, sg, sp, colsum);
    dim3 g2(S_ / 16, B_);
    hdsym_phase2<<<g2, 32, 0, stream>>>(sg, diag);
    hdsym_phase3<<<(B_ * S_ * DM_ + 255) / 256, 256, 0, stream>>>(diag, colsum, sp, out);
}
